// HierarchicalVQ_18468359373178
// MI455X (gfx1250) — matrix operand order repaired and checked by reading, not compiled, not run on hardware
//
#include <hip/hip_runtime.h>
#include <hip/hip_fp16.h>
#include <math.h>

typedef __attribute__((ext_vector_type(16))) _Float16 v16h;
typedef __attribute__((ext_vector_type(8)))  _Float16 v8h;
typedef __attribute__((ext_vector_type(8)))  float    v8f;

#define TILES   8        // M-tiles per block == waves per block
#define CHUNK   128      // codewords staged per LDS chunk
#define CB_K    1024
#define DIM     64
#define HW      4096     // 64*64
#define APAD    72       // padded f16 row stride (144B: 16B-aligned, bank-conflict-free)

// ---------------------------------------------------------------------------
// Zero workspace (counts must start at 0 every launch; graph-replay safe)
// ---------------------------------------------------------------------------
__global__ void vq_zero(unsigned* __restrict__ cnt) {
    int i = blockIdx.x * 256 + threadIdx.x;
    if (i < 4 * CB_K) cnt[i] = 0u;
}

// ---------------------------------------------------------------------------
// Main VQ kernel: scores via f16 WMMA (C preloaded with -0.5*||c||^2 so the
// accumulator IS the argmax score), argmin->argmax, gather, partial SSE.
// grid = (512, 4 layers), block = 256 (8 wave32)
// ---------------------------------------------------------------------------
__global__ __launch_bounds__(256) void vq_main(
    const float* __restrict__ x,
    const float* __restrict__ cb0, const float* __restrict__ cb1,
    const float* __restrict__ cb2, const float* __restrict__ cb3,
    float* __restrict__ outQ, float* __restrict__ outIdx,
    float* __restrict__ wsPartial /*[4*512]*/, unsigned* __restrict__ wsCnt /*[4*1024]*/)
{
    __shared__ _Float16 Ash[TILES * 16 * APAD];  // x tiles, [tile][m][c] f16
    __shared__ _Float16 Csh[CHUNK * APAD];       // codebook chunk, [j][c] f16
    __shared__ float    negh[CHUNK];             // -0.5*||c||^2 (fp32 exact)
    __shared__ int      sIdx[TILES * 16];
    __shared__ float    waveSse[TILES];

    const int l    = blockIdx.y;
    const float* cb = (l == 0) ? cb0 : (l == 1) ? cb1 : (l == 2) ? cb2 : cb3;
    const int tid  = threadIdx.x;
    const int lane = tid & 31;
    const int wid  = tid >> 5;
    const int tile0 = blockIdx.x * TILES;

    // ---- Stage A tiles: coalesced (16 consecutive floats per channel) ----
    for (int i = tid; i < TILES * DIM * 16; i += 256) {
        int m = i & 15, c = (i >> 4) & 63, t = i >> 10;
        int n0 = (tile0 + t) * 16;
        int b = n0 >> 12, hw0 = n0 & 4095;
        float v = x[((b * 256 + l * 64 + c) * HW) + hw0 + m];
        Ash[(t * 16 + m) * APAD + c] = (_Float16)v;
    }

    float bestS[8]; int bestJ[8];
    #pragma unroll
    for (int r = 0; r < 8; ++r) { bestS[r] = -3.4e38f; bestJ[r] = 0; }
    __syncthreads();

    // ---- A fragments for this wave's tile (ISA 16-bit A 16x32 layout) ----
    // lanes 0-15: M=lane, K runs {0..7,16..23}(+32); lanes 16-31: M=lane-16, +8
    const _Float16* arow = &Ash[(wid * 16 + (lane & 15)) * APAD];
    const int ak = (lane < 16) ? 0 : 8;
    v16h a0, a1;
    *((v8h*)&a0)       = *((const v8h*)(arow + ak));
    *(((v8h*)&a0) + 1) = *((const v8h*)(arow + ak + 16));
    *((v8h*)&a1)       = *((const v8h*)(arow + ak + 32));
    *(((v8h*)&a1) + 1) = *((const v8h*)(arow + ak + 48));

    // ---- Sweep codebook in chunks ----
    for (int ch = 0; ch < CB_K / CHUNK; ++ch) {
        // stage f16 codebook chunk (coalesced over full rows)
        for (int i = tid; i < CHUNK * DIM; i += 256) {
            int c = i & 63, j = i >> 6;
            Csh[j * APAD + c] = (_Float16)cb[(ch * CHUNK + j) * DIM + c];
        }
        // exact fp32 -0.5*||c||^2: two threads per codeword + shfl combine
        {
            int j = tid >> 1, half = tid & 1;
            const float* r = cb + (ch * CHUNK + j) * DIM + half * 32;
            float s = 0.f;
            #pragma unroll
            for (int c = 0; c < 32; ++c) { float v = r[c]; s += v * v; }
            s += __shfl_xor(s, 1, 32);
            if (half == 0) negh[j] = -0.5f * s;
        }
        __syncthreads();

        // prefetch next chunk's codebook (32KB: 128B per thread) -> overlaps WMMA
        if (ch + 1 < CB_K / CHUNK)
            __builtin_prefetch(cb + (ch + 1) * CHUNK * DIM + tid * 32, 0, 1);

        for (int ct = 0; ct < CHUNK / 16; ++ct) {
            // B fragments (32x16 f16): lanes 0-15: N=lane, K={0..7,16..23}; lanes 16-31: +8
            const _Float16* brow = &Csh[(ct * 16 + (lane & 15)) * APAD];
            const int bk = (lane < 16) ? 0 : 8;
            v16h b0, b1;
            *((v8h*)&b0)       = *((const v8h*)(brow + bk));
            *(((v8h*)&b0) + 1) = *((const v8h*)(brow + bk + 16));
            *((v8h*)&b1)       = *((const v8h*)(brow + bk + 32));
            *(((v8h*)&b1) + 1) = *((const v8h*)(brow + bk + 48));

            // C preloaded with -0.5*||c||^2 (per-column == per-lane constant)
            const float nh = negh[ct * 16 + (lane & 15)];
            v8f acc;
            #pragma unroll
            for (int r = 0; r < 8; ++r) acc[r] = nh;

            acc = __builtin_amdgcn_wmma_f32_16x16x32_f16(false, a0, false, b0,
                                                         (short)0, acc, false, false);
            acc = __builtin_amdgcn_wmma_f32_16x16x32_f16(false, a1, false, b1,
                                                         (short)0, acc, false, false);

            // score = x.c - 0.5*||c||^2 ; argmin(d) == argmax(score).
            // strict > keeps first (lowest-j) max within a lane: matches argmin.
            const int j = ch * CHUNK + ct * 16 + (lane & 15);
            #pragma unroll
            for (int r = 0; r < 8; ++r) {
                if (acc[r] > bestS[r]) { bestS[r] = acc[r]; bestJ[r] = j; }
            }
        }
        __syncthreads();
    }

    // ---- Cross-lane argmax within each 16-lane group (rows split by half) --
    #pragma unroll
    for (int r = 0; r < 8; ++r) {
        float s = bestS[r]; int j = bestJ[r];
        #pragma unroll
        for (int off = 1; off < 16; off <<= 1) {
            float os = __shfl_xor(s, off, 32);
            int   oj = __shfl_xor(j, off, 32);
            if (os > s || (os == s && oj < j)) { s = os; j = oj; }
        }
        bestS[r] = s; bestJ[r] = j;
    }
    if ((lane & 15) == 0) {
        int mbase = (lane < 16) ? 0 : 8;   // C/D layout: VGPR r -> M = r + 8*(lane/16)
        #pragma unroll
        for (int r = 0; r < 8; ++r) sIdx[wid * 16 + mbase + r] = bestJ[r];
    }
    __syncthreads();

    // ---- Output: exact fp32 gather, coalesced NCHW store, SSE, counts ----
    {
        int n0 = (tile0 + wid) * 16;
        int b = n0 >> 12, hw0 = n0 & 4095;
        int m = lane & 15;
        int myIdx = sIdx[wid * 16 + m];
        const float* qrow = cb + myIdx * DIM;
        const float* xrow = x    + (size_t)(b * 256 + l * 64) * HW + hw0 + m;
        float*       orow = outQ + (size_t)(b * 256 + l * 64) * HW + hw0 + m;
        float sse = 0.f;
        #pragma unroll 8
        for (int k = 0; k < 32; ++k) {
            int c = (lane >> 4) + 2 * k;     // lanes<16: even c, lanes>=16: odd c
            float q  = qrow[c];
            float xv = xrow[(size_t)c * HW];
            orow[(size_t)c * HW] = q;
            float df = q - xv; sse += df * df;
        }
        if (lane < 16) {
            outIdx[(b * 4 + l) * HW + hw0 + m] = (float)myIdx;
            atomicAdd(&wsCnt[l * CB_K + myIdx], 1u);
        }
        #pragma unroll
        for (int off = 1; off < 32; off <<= 1) sse += __shfl_xor(sse, off, 32);
        if (lane == 0) waveSse[wid] = sse;
    }
    __syncthreads();
    if (tid == 0) {                          // deterministic per-block partial
        float s = 0.f;
        #pragma unroll
        for (int w = 0; w < TILES; ++w) s += waveSse[w];
        wsPartial[blockIdx.y * 512 + blockIdx.x] = s;
    }
}

// ---------------------------------------------------------------------------
// Finalize: loss = 1.25 * SSE / (N*DIM); perplexity from histogram
// ---------------------------------------------------------------------------
__global__ void vq_final(const unsigned* __restrict__ cnt,
                         const float* __restrict__ partial,
                         float* __restrict__ outLoss, float* __restrict__ outPerp)
{
    __shared__ float red[256];
    const int l = blockIdx.x, tid = threadIdx.x;

    float s = 0.f;
    for (int i = tid; i < 512; i += 256) s += partial[l * 512 + i];
    red[tid] = s; __syncthreads();
    for (int st = 128; st > 0; st >>= 1) {
        if (tid < st) red[tid] += red[tid + st];
        __syncthreads();
    }
    float sse = red[0];
    __syncthreads();

    float h = 0.f;
    for (int i = tid; i < CB_K; i += 256) {
        float p = (float)cnt[l * CB_K + i] * (1.0f / 65536.0f);
        h += p * logf(p + 1e-10f);
    }
    red[tid] = h; __syncthreads();
    for (int st = 128; st > 0; st >>= 1) {
        if (tid < st) red[tid] += red[tid + st];
        __syncthreads();
    }
    if (tid == 0) {
        outLoss[l] = 1.25f * sse / (float)(65536.0 * 64.0);
        outPerp[l] = expf(-red[0]);
    }
}

// ---------------------------------------------------------------------------
extern "C" void kernel_launch(void* const* d_in, const int* in_sizes, int n_in,
                              void* d_out, int out_size, void* d_ws, size_t ws_size,
                              hipStream_t stream)
{
    const float* x   = (const float*)d_in[0];
    const float* cb0 = (const float*)d_in[1];
    const float* cb1 = (const float*)d_in[2];
    const float* cb2 = (const float*)d_in[3];
    const float* cb3 = (const float*)d_in[4];

    float* out     = (float*)d_out;
    float* outQ    = out;                         // 16*256*64*64 = 16,777,216
    float* outIdx  = out + 16777216;              // 16*4*64*64   =    262,144
    float* outLoss = out + 16777216 + 262144;     // 4
    float* outPerp = outLoss + 4;                 // 4

    unsigned* wsCnt     = (unsigned*)d_ws;                       // 4*1024 u32
    float*    wsPartial = (float*)((char*)d_ws + 4 * CB_K * 4);  // 4*512 f32

    vq_zero<<<16, 256, 0, stream>>>(wsCnt);
    vq_main<<<dim3(512, 4), 256, 0, stream>>>(x, cb0, cb1, cb2, cb3,
                                              outQ, outIdx, wsPartial, wsCnt);
    vq_final<<<4, 256, 0, stream>>>(wsCnt, wsPartial, outLoss, outPerp);
}
